// TemporalAttentionPooling_85890755985425
// MI455X (gfx1250) — matrix operand order repaired and checked by reading, not compiled, not run on hardware
//
#include <hip/hip_runtime.h>
#include <hip/hip_bf16.h>

#define B_ 32
#define T_ 4096
#define D_ 512
#define A_ 64
#define NCH 16            // T chunks for context partial sums
#define TCH (T_ / NCH)    // 256

typedef __attribute__((ext_vector_type(16))) __bf16        v16bf;
typedef __attribute__((ext_vector_type(8)))  float         v8f;
typedef __attribute__((ext_vector_type(8)))  unsigned int  v8u;

__device__ __forceinline__ __bf16 cvt_bf16(float f) {
    // round-to-nearest-even f32 -> bf16 (scalar path for the LDS W1 fill)
    unsigned int u = __builtin_bit_cast(unsigned int, f);
    u += 0x7fffu + ((u >> 16) & 1u);
    unsigned short h = (unsigned short)(u >> 16);
    return __builtin_bit_cast(__bf16, h);
}

// Packed f32 -> 2xbf16 via the gfx1250 hardware convert.
// v_cvt_pk_bf16_f32: D = { bf16(S1) [31:16], bf16(S0) [15:0] }
__device__ __forceinline__ unsigned int pk_bf16(float lo, float hi) {
    unsigned int r;
    asm("v_cvt_pk_bf16_f32 %0, %1, %2" : "=v"(r) : "v"(lo), "v"(hi));
    return r;
}

__device__ __forceinline__ float fast_tanh(float x) {
#if __has_builtin(__builtin_amdgcn_tanhf)
    return __builtin_amdgcn_tanhf(x);                // v_tanh_f32 (CDNA5 TRANS op)
#else
    return tanhf(x);
#endif
}

// Sum across the 16 lanes of each half-wave using ds_swizzle_b32
// (group-of-32 mode: offset = xor_mask<<10 | and_mask(0x1F)).
__device__ __forceinline__ float halfwave_sum16(float s) {
    s += __builtin_bit_cast(float, __builtin_amdgcn_ds_swizzle(
             __builtin_bit_cast(int, s), (1 << 10) | 0x1F));
    s += __builtin_bit_cast(float, __builtin_amdgcn_ds_swizzle(
             __builtin_bit_cast(int, s), (2 << 10) | 0x1F));
    s += __builtin_bit_cast(float, __builtin_amdgcn_ds_swizzle(
             __builtin_bit_cast(int, s), (4 << 10) | 0x1F));
    s += __builtin_bit_cast(float, __builtin_amdgcn_ds_swizzle(
             __builtin_bit_cast(int, s), (8 << 10) | 0x1F));
    return s;
}

__device__ __forceinline__ v8f zero_v8f() {
    v8f z;
#pragma unroll
    for (int e = 0; e < 8; ++e) z[e] = 0.0f;
    return z;
}

// ---------------------------------------------------------------------------
// Kernel 1: masked attention scores via bf16 WMMA.
//   H = tanh(X_tile(16x512) * W1^T(512x64) + b1);  score = H . w2
// Workgroup = 256 threads = 8 waves; each wave owns 16 rows of T.
// Grid = B * (T/128).
// ---------------------------------------------------------------------------
__global__ __launch_bounds__(256) void scores_kernel(
    const float* __restrict__ x, const unsigned char* __restrict__ mask,
    const float* __restrict__ W1, const float* __restrict__ b1,
    const float* __restrict__ w2, float* __restrict__ scores /* B*T */) {

    __shared__ __bf16 sW1[A_ * D_];  // 64 KB, row-major [a][d], bf16
    const int tid = threadIdx.x;
    for (int i = tid; i < A_ * D_; i += 256) sW1[i] = cvt_bf16(W1[i]);
    __syncthreads();

    const int blocksPerB = T_ / 128;
    const int b    = blockIdx.x / blocksPerB;
    const int t0wg = (blockIdx.x % blocksPerB) * 128;
    const int wave = tid >> 5;
    const int lane = tid & 31;
    const int n    = lane & 15;   // column / M-row selector within half-wave
    const int hi   = lane >> 4;   // half-wave: selects K sub-range
    const int t0   = t0wg + wave * 16;

    // A-matrix: lane's M row = t0 + n
    const float* xrow = x + ((size_t)b * T_ + (size_t)(t0 + n)) * D_;

    v8f acc[4];
#pragma unroll
    for (int j = 0; j < 4; ++j) acc[j] = zero_v8f();

    for (int k0 = 0; k0 < D_; k0 += 32) {
        // A-fragment (16x32 bf16) per ISA layout:
        //   halves 0..7  -> K = k0 + hi*8 + 0..7
        //   halves 8..15 -> K = k0 + 16 + hi*8 + 0..7
        const float4* p0 = (const float4*)(xrow + k0 + hi * 8);
        float4 f0 = p0[0], f1 = p0[1];
        const float4* p1 = (const float4*)(xrow + k0 + 16 + hi * 8);
        float4 f2 = p1[0], f3 = p1[1];
        v8u pk;
        pk[0] = pk_bf16(f0.x, f0.y);
        pk[1] = pk_bf16(f0.z, f0.w);
        pk[2] = pk_bf16(f1.x, f1.y);
        pk[3] = pk_bf16(f1.z, f1.w);
        pk[4] = pk_bf16(f2.x, f2.y);
        pk[5] = pk_bf16(f2.z, f2.w);
        pk[6] = pk_bf16(f3.x, f3.y);
        pk[7] = pk_bf16(f3.z, f3.w);
        v16bf a = __builtin_bit_cast(v16bf, pk);

#pragma unroll
        for (int j = 0; j < 4; ++j) {
            // B-fragment (32x16 bf16): B[k][nn] = W1[j*16+nn][k0+k]
            // lane holds column n, halves i -> K = k0 + hi*8 + i (i<8), k0 + 16 + hi*8 + i-8 (i>=8)
            const __bf16* wp = &sW1[(j * 16 + n) * D_ + k0 + hi * 8];
            v16bf bb;
#pragma unroll
            for (int i = 0; i < 16; ++i) bb[i] = wp[i < 8 ? i : i + 8];
            acc[j] = __builtin_amdgcn_wmma_f32_16x16x32_bf16(
                false, a, false, bb, (short)0, acc[j], false, false);
        }
    }

    // Epilogue: s[t] = sum_a w2[a] * tanh(acc + b1[a]); reduce over N lanes.
    float w2v[4], b1v[4];
#pragma unroll
    for (int j = 0; j < 4; ++j) {
        int aI = j * 16 + n;
        w2v[j] = w2[aI];
        b1v[j] = b1[aI];
    }

#pragma unroll
    for (int v = 0; v < 8; ++v) {
        float s = 0.0f;
#pragma unroll
        for (int j = 0; j < 4; ++j)
            s += w2v[j] * fast_tanh(acc[j][v] + b1v[j]);
        s = halfwave_sum16(s);   // sum over the 16 N-lanes of this half-wave
        if (n == 0) {
            int t = t0 + v + hi * 8;              // C/D layout: M = v + 8*hi
            size_t idx = (size_t)b * T_ + t;
            scores[idx] = mask[idx] ? s : -__builtin_inff();
        }
    }
}

// ---------------------------------------------------------------------------
// Kernel 2: softmax over T per batch row, in place. Grid = B, block = 256.
// ---------------------------------------------------------------------------
__global__ __launch_bounds__(256) void softmax_kernel(float* __restrict__ w) {
    const int b = blockIdx.x;
    const int tid = threadIdx.x;
    const int wave = tid >> 5, lane = tid & 31;
    __shared__ float red[8];

    float v[T_ / 256];
    float m = -__builtin_inff();
#pragma unroll
    for (int i = 0; i < T_ / 256; ++i) {
        v[i] = w[(size_t)b * T_ + tid + i * 256];
        m = fmaxf(m, v[i]);
    }
#pragma unroll
    for (int d = 1; d < 32; d <<= 1) m = fmaxf(m, __shfl_xor(m, d));
    if (lane == 0) red[wave] = m;
    __syncthreads();
#pragma unroll
    for (int k = 0; k < 8; ++k) m = fmaxf(m, red[k]);

    float s = 0.0f;
#pragma unroll
    for (int i = 0; i < T_ / 256; ++i) {
        float e = __expf(v[i] - m);   // exp(-inf) == 0 for masked steps
        v[i] = e;
        s += e;
    }
#pragma unroll
    for (int d = 1; d < 32; d <<= 1) s += __shfl_xor(s, d);
    __syncthreads();
    if (lane == 0) red[wave] = s;
    __syncthreads();
    s = 0.0f;
#pragma unroll
    for (int k = 0; k < 8; ++k) s += red[k];

    float inv = 1.0f / s;
#pragma unroll
    for (int i = 0; i < T_ / 256; ++i)
        w[(size_t)b * T_ + tid + i * 256] = v[i] * inv;
}

// ---------------------------------------------------------------------------
// Kernel 3: partial context: part[b,ch,d] = sum_{t in chunk} w[b,t]*x[b,t,d]
// Grid = B*NCH, block = 256 (each thread owns 2 of D=512 columns).
// ---------------------------------------------------------------------------
__global__ __launch_bounds__(256) void context_partial_kernel(
    const float* __restrict__ x, const float* __restrict__ w,
    float* __restrict__ part) {
    const int b  = blockIdx.x / NCH;
    const int ch = blockIdx.x % NCH;
    const int d0 = threadIdx.x;
    const float* xb = x + ((size_t)b * T_ + (size_t)ch * TCH) * D_;
    const float* wb = w + (size_t)b * T_ + (size_t)ch * TCH;
    float a0 = 0.0f, a1 = 0.0f;
    for (int t = 0; t < TCH; ++t) {
        float wt = wb[t];
        a0 = fmaf(wt, xb[(size_t)t * D_ + d0], a0);
        a1 = fmaf(wt, xb[(size_t)t * D_ + d0 + 256], a1);
    }
    float* pp = part + (size_t)blockIdx.x * D_;
    pp[d0] = a0;
    pp[d0 + 256] = a1;
}

// ---------------------------------------------------------------------------
// Kernel 4: reduce chunk partials -> context. Grid = B, block = 256.
// ---------------------------------------------------------------------------
__global__ __launch_bounds__(256) void context_reduce_kernel(
    const float* __restrict__ part, float* __restrict__ ctx) {
    const int b = blockIdx.x;
    const int d0 = threadIdx.x;
    float a0 = 0.0f, a1 = 0.0f;
    for (int ch = 0; ch < NCH; ++ch) {
        const float* pp = part + ((size_t)b * NCH + ch) * D_;
        a0 += pp[d0];
        a1 += pp[d0 + 256];
    }
    ctx[(size_t)b * D_ + d0] = a0;
    ctx[(size_t)b * D_ + d0 + 256] = a1;
}

// ---------------------------------------------------------------------------
extern "C" void kernel_launch(void* const* d_in, const int* in_sizes, int n_in,
                              void* d_out, int out_size, void* d_ws, size_t ws_size,
                              hipStream_t stream) {
    (void)in_sizes; (void)n_in; (void)out_size; (void)ws_size;
    const float*         x    = (const float*)d_in[0];
    const unsigned char* mask = (const unsigned char*)d_in[1];  // numpy bool = 1 byte
    const float*         W1   = (const float*)d_in[2];
    const float*         b1   = (const float*)d_in[3];
    const float*         w2   = (const float*)d_in[4];

    float* ctx     = (float*)d_out;          // context: B*D floats (first in tuple)
    float* weights = ctx + B_ * D_;          // weights: B*T floats (second)
    float* part    = (float*)d_ws;           // B*NCH*D floats = 1 MB scratch

    scores_kernel<<<dim3(B_ * (T_ / 128)), dim3(256), 0, stream>>>(
        x, mask, W1, b1, w2, weights);
    softmax_kernel<<<dim3(B_), dim3(256), 0, stream>>>(weights);
    context_partial_kernel<<<dim3(B_ * NCH), dim3(256), 0, stream>>>(
        x, weights, part);
    context_reduce_kernel<<<dim3(B_), dim3(256), 0, stream>>>(part, ctx);
}
